// LSTM_Generator_85633057948170
// MI455X (gfx1250) — matrix operand order repaired and checked by reading, not compiled, not run on hardware
//
#include <hip/hip_runtime.h>

// ---------------- problem constants (match reference) ----------------
#define IMG_STEPS 64
#define GEN_EMB   100
#define HDIM      128
#define IN0_K     356          // 2H + GEN_EMB
#define STR_Y     264          // 256 + 8 pad (bf16 elems) -> 528B row stride, 16B aligned
#define STR_X     136          // 128 + 8 pad

// ---------------- CDNA5 WMMA types ----------------
typedef __attribute__((ext_vector_type(16))) __bf16 v16bf;
typedef __attribute__((ext_vector_type(8)))  float  v8f;
typedef __attribute__((ext_vector_type(4)))  unsigned int u32x4;

union Frag { v16bf v; u32x4 q[2]; };
union V8   { v8f v; float f[8]; };

// ---------------- packed-weight workspace layout (u16 elements) ----------------
constexpr size_t SZ_WIH0 = (size_t)32 * 12 * 512;  // 512x384(pad) -> 196608
constexpr size_t SZ_WHH  = (size_t)32 * 4  * 512;  // 512x128     ->  65536
constexpr size_t SZ_WIH1 = (size_t)32 * 8  * 512;  // 512x256     -> 131072
constexpr size_t SZ_FC1  = (size_t)8  * 8  * 512;  // 128x256     ->  32768
constexpr size_t SZ_FC2  = (size_t)8  * 4  * 512;  // 128x128     ->  16384

constexpr size_t OFF_WIH0F = 0;
constexpr size_t OFF_WHH0F = OFF_WIH0F + SZ_WIH0;
constexpr size_t OFF_WIH0B = OFF_WHH0F + SZ_WHH;
constexpr size_t OFF_WHH0B = OFF_WIH0B + SZ_WIH0;
constexpr size_t OFF_WIH1F = OFF_WHH0B + SZ_WHH;
constexpr size_t OFF_WHH1F = OFF_WIH1F + SZ_WIH1;
constexpr size_t OFF_WIH1B = OFF_WHH1F + SZ_WHH;
constexpr size_t OFF_WHH1B = OFF_WIH1B + SZ_WIH1;
constexpr size_t OFF_FC1   = OFF_WHH1B + SZ_WHH;
constexpr size_t OFF_FC2   = OFF_FC1   + SZ_FC1;
constexpr size_t W_TOTAL   = OFF_FC2   + SZ_FC2;   // 966656 elems = 1.93 MB

// bias ws layout (f32, after weights): [0:512) c0f  [512:1024) c0b
// [1024:1536) c1f  [1536:2048) c1b  [2048:2176) fc1_b  [2176:2304) fc2_b

// ---------------- helpers ----------------
__device__ __forceinline__ unsigned short f2bf(float x) {
  unsigned u = __float_as_uint(x);
  u += 0x7FFFu + ((u >> 16) & 1u);           // round-to-nearest-even
  return (unsigned short)(u >> 16);
}
__device__ __forceinline__ float sigm(float x) { return 1.0f / (1.0f + __expf(-x)); }
__device__ __forceinline__ void zero8(V8& a) {
#pragma unroll
  for (int r = 0; r < 8; ++r) a.f[r] = 0.0f;
}

// A fragment (16x32 bf16): lane holds row M=lane&15; halves 0..7 = K[b..b+8),
// halves 8..15 = K[16+b..16+b+8), b = (lane<16 ? 0 : 8).  Two 16B LDS reads.
__device__ __forceinline__ void loadA(Frag& a, const unsigned short* base,
                                      int strideElems, int col0, int lane) {
  const int m = lane & 15;
  const int b = (lane < 16) ? 0 : 8;
  const unsigned short* r = base + m * strideElems + col0 + b;
  a.q[0] = *(const u32x4*)(r);
  a.q[1] = *(const u32x4*)(r + 16);
}

// B fragment (32x16 bf16), packed so lane reads its 32 contiguous bytes.
__device__ __forceinline__ void loadB(Frag& b, const unsigned short* blk, int lane) {
  const unsigned short* p = blk + lane * 16;
  b.q[0] = *(const u32x4*)(p);
  b.q[1] = *(const u32x4*)(p + 8);
}

// ---- GEMM over nK K-chunks into the 4 gate accumulators (tiles t*8 + wv) ----
// A comes from LDS (aBase, aStride, starting col aCol0); B from packed weights,
// weight k-chunk index = kcOff + i.  Unroll 2: the scheduler issues iteration
// i+1's loads ahead of iteration i's WMMAs (cross-iteration renaming, no movs)
// while keeping the live set ~2x40 + accs well under 256 VGPRs (no spills).
__device__ __forceinline__ void gemm4(V8* acc,
                                      const unsigned short* aBase, int aStride, int aCol0,
                                      int nK, const unsigned short* wmat, int kch, int kcOff,
                                      int wv, int lane) {
#pragma unroll 2
  for (int i = 0; i < nK; ++i) {
    Frag a;
    loadA(a, aBase, aStride, aCol0 + i * 32, lane);
    Frag b4[4];
#pragma unroll
    for (int t = 0; t < 4; ++t)
      loadB(b4[t], wmat + ((size_t)((t * 8 + wv) * kch + kcOff + i)) * 512, lane);
#pragma unroll
    for (int t = 0; t < 4; ++t)
      acc[t].v = __builtin_amdgcn_wmma_f32_16x16x32_bf16(
          false, a.v, false, b4[t].v, (short)0, acc[t].v, false, false);
  }
}

// ---- single-tile GEMM (fc layers): output tile = wv ----
__device__ __forceinline__ void gemm1(V8& acc,
                                      const unsigned short* aBase, int aStride,
                                      int nK, const unsigned short* wmat, int kch,
                                      int wv, int lane) {
#pragma unroll 2
  for (int i = 0; i < nK; ++i) {
    Frag a;
    loadA(a, aBase, aStride, i * 32, lane);
    Frag b;
    loadB(b, wmat + ((size_t)(wv * kch + i)) * 512, lane);
    acc.v = __builtin_amdgcn_wmma_f32_16x16x32_bf16(
        false, a.v, false, b.v, (short)0, acc.v, false, false);
  }
}

// LSTM pointwise: c = sig(f)*c + sig(i)*tanh(g); h = sig(o)*tanh(c); h -> LDS bf16
__device__ __forceinline__ void lstm_update_store(V8* acc, V8& c,
                                                  const float* __restrict__ bias, int bOff,
                                                  unsigned short* dst, int dcol0,
                                                  int lane, int wv) {
  const int n  = lane & 15;
  const int mb = (lane < 16) ? 0 : 8;
  const float bi = bias[bOff + 0 * HDIM + wv * 16 + n];
  const float bf = bias[bOff + 1 * HDIM + wv * 16 + n];
  const float bg = bias[bOff + 2 * HDIM + wv * 16 + n];
  const float bo = bias[bOff + 3 * HDIM + wv * 16 + n];
#pragma unroll
  for (int r = 0; r < 8; ++r) {
    const float iv = sigm(acc[0].f[r] + bi);
    const float fv = sigm(acc[1].f[r] + bf);
    const float gv = tanhf(acc[2].f[r] + bg);
    const float ov = sigm(acc[3].f[r] + bo);
    const float cv = fv * c.f[r] + iv * gv;
    c.f[r] = cv;
    dst[(mb + r) * STR_Y + dcol0 + n] = f2bf(ov * tanhf(cv));
  }
}

// ---------------- prep kernels ----------------
// Repack W (NxK row-major f32) into bf16 B-fragment blocks:
// block(ntile,kc) element [lane*16+j] = W[ntile*16+(lane&15)][kc*32+8*(lane>>4)+16*(j>>3)+(j&7)], zero-pad K.
__global__ void pack_weight_kernel(const float* __restrict__ src,
                                   unsigned short* __restrict__ dst,
                                   int K, int KCH, int total) {
  const int tid = blockIdx.x * blockDim.x + threadIdx.x;
  if (tid >= total) return;
  const int idx   = tid & 511;
  const int blk   = tid >> 9;
  const int kc    = blk % KCH;
  const int ntile = blk / KCH;
  const int lane  = idx >> 4;
  const int j     = idx & 15;
  const int k = kc * 32 + 8 * (lane >> 4) + 16 * (j >> 3) + (j & 7);
  const int n = ntile * 16 + (lane & 15);
  const float v = (k < K) ? src[(size_t)n * K + k] : 0.0f;
  dst[tid] = f2bf(v);
}

__global__ void prep_bias_kernel(const float* bi0f, const float* bh0f,
                                 const float* bi0b, const float* bh0b,
                                 const float* bi1f, const float* bh1f,
                                 const float* bi1b, const float* bh1b,
                                 const float* fc1b, const float* fc2b,
                                 float* __restrict__ dst) {
  const int t = blockIdx.x * blockDim.x + threadIdx.x;
  if      (t < 512)  dst[t] = bi0f[t]        + bh0f[t];
  else if (t < 1024) dst[t] = bi0b[t - 512]  + bh0b[t - 512];
  else if (t < 1536) dst[t] = bi1f[t - 1024] + bh1f[t - 1024];
  else if (t < 2048) dst[t] = bi1b[t - 1536] + bh1b[t - 1536];
  else if (t < 2176) dst[t] = fc1b[t - 2048];
  else if (t < 2304) dst[t] = fc2b[t - 2176];
}

// ---------------- main kernel: 16 batch rows per workgroup, 8 waves ----------------
__global__ __launch_bounds__(256) void lstm_gen_kernel(
    const float* __restrict__ x, const unsigned short* __restrict__ w,
    const float* __restrict__ bias, float* __restrict__ out) {
  __shared__ unsigned short y0[2][16 * STR_Y];  // [h0f | h0b] ping-pong
  __shared__ unsigned short h1[2][16 * STR_Y];  // [h1f | h1b] ping-pong (= out_lstm)
  __shared__ unsigned short xs[16 * STR_X];     // embedding, cols 0..99, zero pad
  __shared__ unsigned short ts[16 * STR_X];     // tanh(fc1) buffer

  const int tid  = threadIdx.x;
  const int lane = tid & 31;
  const int wv   = tid >> 5;               // wave 0..7 : owns output cols [wv*16, wv*16+16)
  const int row0 = blockIdx.x * 16;

  for (int i = tid; i < 16 * STR_Y; i += 256) {
    y0[0][i] = 0; y0[1][i] = 0; h1[0][i] = 0; h1[1][i] = 0;
  }
  for (int i = tid; i < 16 * STR_X; i += 256) {
    const int r = i / STR_X, c0 = i % STR_X;
    const float v = (c0 < GEN_EMB) ? x[(size_t)(row0 + r) * GEN_EMB + c0] : 0.0f;
    xs[i] = f2bf(v);
    ts[i] = 0;
  }
  __syncthreads();

  V8 c0f, c0b, c1f, c1b;                   // cell state lives in VGPRs across all steps
  zero8(c0f); zero8(c0b); zero8(c1f); zero8(c1b);

#pragma unroll 1
  for (int step = 0; step < IMG_STEPS; ++step) {
    const int p = step & 1;                // read buffers
    const int q = p ^ 1;                   // write buffers

    // ---- layer 0 forward: G = [out_lstm_prev|x] @ Wih^T + h0f @ Whh^T ----
    {
      V8 acc[4]; zero8(acc[0]); zero8(acc[1]); zero8(acc[2]); zero8(acc[3]);
      gemm4(acc, h1[p], STR_Y, 0, 8, w + OFF_WIH0F, 12, 0, wv, lane);  // out_lstm_prev
      gemm4(acc, xs,    STR_X, 0, 4, w + OFF_WIH0F, 12, 8, wv, lane);  // embedding (+pad)
      gemm4(acc, y0[p], STR_Y, 0, 4, w + OFF_WHH0F,  4, 0, wv, lane);  // h0f prev
      lstm_update_store(acc, c0f, bias, 0, y0[q], wv * 16, lane, wv);
    }
    // ---- layer 0 backward ----
    {
      V8 acc[4]; zero8(acc[0]); zero8(acc[1]); zero8(acc[2]); zero8(acc[3]);
      gemm4(acc, h1[p], STR_Y,   0, 8, w + OFF_WIH0B, 12, 0, wv, lane);
      gemm4(acc, xs,    STR_X,   0, 4, w + OFF_WIH0B, 12, 8, wv, lane);
      gemm4(acc, y0[p], STR_Y, 128, 4, w + OFF_WHH0B,  4, 0, wv, lane); // h0b prev
      lstm_update_store(acc, c0b, bias, 512, y0[q], 128 + wv * 16, lane, wv);
    }
    __syncthreads();

    // ---- layer 1 forward: G = y0_new @ Wih^T + h1f @ Whh^T ----
    {
      V8 acc[4]; zero8(acc[0]); zero8(acc[1]); zero8(acc[2]); zero8(acc[3]);
      gemm4(acc, y0[q], STR_Y, 0, 8, w + OFF_WIH1F, 8, 0, wv, lane);
      gemm4(acc, h1[p], STR_Y, 0, 4, w + OFF_WHH1F, 4, 0, wv, lane);   // h1f prev
      lstm_update_store(acc, c1f, bias, 1024, h1[q], wv * 16, lane, wv);
    }
    // ---- layer 1 backward ----
    {
      V8 acc[4]; zero8(acc[0]); zero8(acc[1]); zero8(acc[2]); zero8(acc[3]);
      gemm4(acc, y0[q], STR_Y,   0, 8, w + OFF_WIH1B, 8, 0, wv, lane);
      gemm4(acc, h1[p], STR_Y, 128, 4, w + OFF_WHH1B, 4, 0, wv, lane); // h1b prev
      lstm_update_store(acc, c1b, bias, 1536, h1[q], 128 + wv * 16, lane, wv);
    }
    __syncthreads();

    // ---- fc1: tanh(out_lstm @ fc1_w^T + b) -> ts (bf16) ----
    {
      V8 acc; zero8(acc);
      gemm1(acc, h1[q], STR_Y, 8, w + OFF_FC1, 8, wv, lane);
      const int n = lane & 15, mb = (lane < 16) ? 0 : 8;
      const float bb = bias[2048 + wv * 16 + n];
#pragma unroll
      for (int r = 0; r < 8; ++r)
        ts[(mb + r) * STR_X + wv * 16 + n] = f2bf(tanhf(acc.f[r] + bb));
    }
    __syncthreads();

    // ---- fc2: tanh(ts @ fc2_w^T + b) -> global out (B,1,64,128) ----
    {
      V8 acc; zero8(acc);
      gemm1(acc, ts, STR_X, 4, w + OFF_FC2, 4, wv, lane);
      const int n = lane & 15, mb = (lane < 16) ? 0 : 8;
      const float bb = bias[2176 + wv * 16 + n];
#pragma unroll
      for (int r = 0; r < 8; ++r)
        out[(size_t)(row0 + mb + r) * (IMG_STEPS * HDIM) + step * HDIM + wv * 16 + n] =
            tanhf(acc.f[r] + bb);
    }
    __syncthreads();   // protects ts and the ping-pong buffers for the next step
  }
}

// ---------------- host entry ----------------
extern "C" void kernel_launch(void* const* d_in, const int* in_sizes, int n_in,
                              void* d_out, int out_size, void* d_ws, size_t ws_size,
                              hipStream_t stream) {
  const float* x      = (const float*)d_in[0];
  // d_in[1] = labels (unused by the reference computation)
  const float* w_ih0f = (const float*)d_in[2];
  const float* w_hh0f = (const float*)d_in[3];
  const float* b_ih0f = (const float*)d_in[4];
  const float* b_hh0f = (const float*)d_in[5];
  const float* w_ih0b = (const float*)d_in[6];
  const float* w_hh0b = (const float*)d_in[7];
  const float* b_ih0b = (const float*)d_in[8];
  const float* b_hh0b = (const float*)d_in[9];
  const float* w_ih1f = (const float*)d_in[10];
  const float* w_hh1f = (const float*)d_in[11];
  const float* b_ih1f = (const float*)d_in[12];
  const float* b_hh1f = (const float*)d_in[13];
  const float* w_ih1b = (const float*)d_in[14];
  const float* w_hh1b = (const float*)d_in[15];
  const float* b_ih1b = (const float*)d_in[16];
  const float* b_hh1b = (const float*)d_in[17];
  const float* fc1_w  = (const float*)d_in[18];
  const float* fc1_b  = (const float*)d_in[19];
  const float* fc2_w  = (const float*)d_in[20];
  const float* fc2_b  = (const float*)d_in[21];

  unsigned short* wp = (unsigned short*)d_ws;        // needs ~1.95 MB of ws
  float* biasws = (float*)(wp + W_TOTAL);            // byte offset is 16-aligned

  const int Bn = in_sizes[0] / GEN_EMB;              // 2048

  auto pack = [&](const float* src, size_t off, int N, int K, int KCH) {
    const int total = (N / 16) * KCH * 512;
    pack_weight_kernel<<<dim3((total + 255) / 256), dim3(256), 0, stream>>>(
        src, wp + off, K, KCH, total);
  };
  pack(w_ih0f, OFF_WIH0F, 512, IN0_K, 12);
  pack(w_hh0f, OFF_WHH0F, 512, 128, 4);
  pack(w_ih0b, OFF_WIH0B, 512, IN0_K, 12);
  pack(w_hh0b, OFF_WHH0B, 512, 128, 4);
  pack(w_ih1f, OFF_WIH1F, 512, 256, 8);
  pack(w_hh1f, OFF_WHH1F, 512, 128, 4);
  pack(w_ih1b, OFF_WIH1B, 512, 256, 8);
  pack(w_hh1b, OFF_WHH1B, 512, 128, 4);
  pack(fc1_w,  OFF_FC1,   128, 256, 8);
  pack(fc2_w,  OFF_FC2,   128, 128, 4);

  prep_bias_kernel<<<dim3(9), dim3(256), 0, stream>>>(
      b_ih0f, b_hh0f, b_ih0b, b_hh0b, b_ih1f, b_hh1f, b_ih1b, b_hh1b,
      fc1_b, fc2_b, biasws);

  lstm_gen_kernel<<<dim3(Bn / 16), dim3(256), 0, stream>>>(
      x, wp, biasws, (float*)d_out);

  (void)n_in; (void)out_size; (void)ws_size;
}
